// Coords2Grid_58213986730418
// MI455X (gfx1250) — matrix operand order repaired and checked by reading, not compiled, not run on hardware
//
#include <hip/hip_runtime.h>
#include <hip/hip_bf16.h>

typedef __attribute__((ext_vector_type(16))) _Float16 v16h;
typedef __attribute__((ext_vector_type(8)))  _Float16 v8h;
typedef __attribute__((ext_vector_type(8)))  float    v8f;

#define G_      48
#define GG_     (G_ * G_)
#define GGG_    (G_ * G_ * G_)
#define NA_     800
#define NT_     28
#define TSTRIDE 808                      // halves per type-row in LDS (bank spread)
#define KCHUNKS (NA_ / 32)
#define RES_    0.5f
#define HALFDIM 11.75f

// gauss = exp(-2 dr2) = exp2(dr2 * -2*log2(e))
#define NEG2_LOG2E (-2.885390081777927f)
// quad = e^-2 (2dr-3)^2 = (3/e - (2/e) dr)^2, clamped at dr=1.5 via max(.,0)
#define QM (-0.73575888234288467f)       // -2/e
#define QB ( 1.1036383235143270f)        //  3/e

__global__ __launch_bounds__(256) void coords2grid_wmma(
    const float* __restrict__ center,
    const float* __restrict__ coords,
    const float* __restrict__ types,
    const float* __restrict__ radii,
    float* __restrict__ out)
{
    // Quadratic-form atom data: dr2 = dot((px,py,pz,pp), q) + qw
    //   q  = (-2 ir2 ax, -2 ir2 ay, -2 ir2 az, ir2),  qw = ir2*|a|^2
    __shared__ float4   s_atomQ[NA_];              // 12.8 KB
    __shared__ float    s_atomW[NA_];              //  3.2 KB
    __shared__ _Float16 s_typeT[NT_ * TSTRIDE];    // transposed f16 [t][a], 45.2 KB

    const int tid = threadIdx.x;

    // ---- stage atoms ----
    for (int a = tid; a < NA_; a += 256) {
        const float ax = coords[3 * a + 0];
        const float ay = coords[3 * a + 1];
        const float az = coords[3 * a + 2];
        const float r  = radii[a];
        const float ir2 = __builtin_amdgcn_rcpf(r * r);
        s_atomQ[a] = make_float4(-2.0f * ir2 * ax, -2.0f * ir2 * ay,
                                 -2.0f * ir2 * az, ir2);
        s_atomW[a] = ir2 * fmaf(ax, ax, fmaf(ay, ay, az * az));
    }
    // ---- stage types transposed as f16: 28 floats/atom = 7 float4 ----
    const float4* types4 = (const float4*)types;
    for (int u = tid; u < NA_ * (NT_ / 4); u += 256) {
        const int a = u / (NT_ / 4);
        const int q = u % (NT_ / 4);
        const float4 v = types4[u];
        const int t0 = q * 4;
        s_typeT[(t0 + 0) * TSTRIDE + a] = (_Float16)v.x;
        s_typeT[(t0 + 1) * TSTRIDE + a] = (_Float16)v.y;
        s_typeT[(t0 + 2) * TSTRIDE + a] = (_Float16)v.z;
        s_typeT[(t0 + 3) * TSTRIDE + a] = (_Float16)v.w;
    }
    __syncthreads();

    const int lane   = tid & 31;
    const int wave   = tid >> 5;
    const int laneHi = lane >> 4;   // 0 or 1 (lane half, per WMMA layouts)
    const int mcol   = lane & 15;   // A: M row; B/C/D: N column

    const int tile = blockIdx.x * 8 + wave;   // 16-voxel tile, 6912 total
    const int vox  = tile * 16 + mcol;        // this lane's M-row voxel
    const int x = vox / GG_;
    const int y = (vox / G_) % G_;
    const int z = vox % G_;

    const float px = (center[0] - HALFDIM) + (float)x * RES_;
    const float py = (center[1] - HALFDIM) + (float)y * RES_;
    const float pz = (center[2] - HALFDIM) + (float)z * RES_;
    const float pp = fmaf(px, px, fmaf(py, py, pz * pz));   // |p|^2, lane constant

    v8f acc0 = {};   // types  0..15
    v8f acc1 = {};   // types 12..27 (12..15 overlap acc0 with identical values)

    const _Float16* tRow0 = s_typeT + (size_t)mcol * TSTRIDE;         // t = mcol
    const _Float16* tRow1 = s_typeT + (size_t)(12 + mcol) * TSTRIDE;  // t = 12+mcol

    for (int c = 0; c < KCHUNKS; ++c) {
        const int kb = c * 32;

        // ---- A fragment: 16x32 f16 densities, contiguous K = e + 16*laneHi ----
        v16h A;
#pragma unroll
        for (int e = 0; e < 16; ++e) {
            const int K = e + (laneHi << 4);
            const float4 q  = s_atomQ[kb + K];
            const float  qw = s_atomW[kb + K];
            // dr2 = d^2/r^2 as a 4-FMA quadratic form
            float dr2 = fmaf(pp, q.w,
                        fmaf(px, q.x, fmaf(py, q.y, fmaf(pz, q.z, qw))));
            dr2 = fmaxf(dr2, 0.0f);                     // FMA cancellation guard
            const float dr    = __builtin_amdgcn_sqrtf(dr2);
            const float gauss = __builtin_amdgcn_exp2f(dr2 * NEG2_LOG2E);
            float u = fmaf(dr, QM, QB);                 // (3 - 2dr)/e
            u = fmaxf(u, 0.0f);                         // folds dr<1.5 cutoff
            const float quad = u * u;
            const float dens = (dr < 1.0f) ? gauss : quad;
            A[e] = (_Float16)dens;
        }

        // ---- B fragments: element e <-> K = e + 16*laneHi (contiguous) ----
        const int ko = kb + (laneHi << 4);
        const v8h b0lo = *(const v8h*)(tRow0 + ko);
        const v8h b0hi = *(const v8h*)(tRow0 + ko + 8);
        const v8h b1lo = *(const v8h*)(tRow1 + ko);
        const v8h b1hi = *(const v8h*)(tRow1 + ko + 8);
        const v16h B0 = __builtin_shufflevector(b0lo, b0hi,
            0, 1, 2, 3, 4, 5, 6, 7, 8, 9, 10, 11, 12, 13, 14, 15);
        const v16h B1 = __builtin_shufflevector(b1lo, b1hi,
            0, 1, 2, 3, 4, 5, 6, 7, 8, 9, 10, 11, 12, 13, 14, 15);

        acc0 = __builtin_amdgcn_wmma_f32_16x16x32_f16(
            false, A, false, B0, (short)0, acc0, false, false);
        acc1 = __builtin_amdgcn_wmma_f32_16x16x32_f16(
            false, A, false, B1, (short)0, acc1, false, false);
    }

    // ---- store: C/D layout VGPR i -> M = 8*laneHi + i, N = mcol ----
    // Each lane owns 8 consecutive voxels: two b128 stores per accumulator.
    const int mBase = tile * 16 + laneHi * 8;
    float* p0 = out + (size_t)mcol * GGG_ + mBase;         // t = mcol
    float* p1 = out + (size_t)(12 + mcol) * GGG_ + mBase;  // t = 12+mcol
    *(float4*)(p0)     = make_float4(acc0[0], acc0[1], acc0[2], acc0[3]);
    *(float4*)(p0 + 4) = make_float4(acc0[4], acc0[5], acc0[6], acc0[7]);
    *(float4*)(p1)     = make_float4(acc1[0], acc1[1], acc1[2], acc1[3]);
    *(float4*)(p1 + 4) = make_float4(acc1[4], acc1[5], acc1[6], acc1[7]);
}

extern "C" void kernel_launch(void* const* d_in, const int* in_sizes, int n_in,
                              void* d_out, int out_size, void* d_ws, size_t ws_size,
                              hipStream_t stream) {
    (void)in_sizes; (void)n_in; (void)out_size; (void)d_ws; (void)ws_size;
    const float* center = (const float*)d_in[0];
    const float* coords = (const float*)d_in[1];
    const float* types  = (const float*)d_in[2];
    const float* radii  = (const float*)d_in[3];
    float* out = (float*)d_out;

    // 48^3 / 16 voxels-per-tile = 6912 tiles; 8 waves per 256-thread block
    const int nBlocks = (GGG_ / 16) / 8;   // 864
    coords2grid_wmma<<<dim3(nBlocks), dim3(256), 0, stream>>>(
        center, coords, types, radii, out);
}
